// STGraphTransformer_17437567221887
// MI455X (gfx1250) — compile-verified, matrix operand order checked by reading, not run on hardware
//
#include <hip/hip_runtime.h>
#include <hip/hip_bf16.h>
#include <math.h>

// ---------------- problem constants ----------------
#define B_    2
#define N_    10000
#define T1_   13
#define SD_   8
#define TED_  16
#define CIN_  24
#define XIN_  (CIN_ + TED_)   // 40
#define H_    128
#define L_    4
#define NH_   4
#define FF_   256
#define E_    160000
#define EN_   (E_ + N_)       // edges + self loops
#define TOK_  (B_ * N_ * T1_) // 260000
#define EPS_  1e-5f

#define MT_   80              // M tile: 5 x 16 sub-tiles (260000 % 80 == 0)
#define MSUB_ 5

typedef __attribute__((ext_vector_type(16))) __bf16 v16bf;
typedef __attribute__((ext_vector_type(8)))  __bf16 v8bf;
typedef __attribute__((ext_vector_type(8)))  float  v8f;
typedef __attribute__((ext_vector_type(4)))  unsigned int v4u;
typedef __attribute__((ext_vector_type(8)))  int    v8i;
typedef __attribute__((ext_vector_type(4)))  int    v4i;
typedef __attribute__((address_space(3)))    float  lds_f32;

__device__ __forceinline__ void wait_asynccnt0()
{
#if __has_builtin(__builtin_amdgcn_s_wait_asynccnt)
    __builtin_amdgcn_s_wait_asynccnt(0);
#else
    asm volatile("s_wait_asynccnt 0" ::: "memory");
#endif
}

#if __has_builtin(__builtin_amdgcn_tensor_load_to_lds)
// 2D TDM load: tile (d0_elems x d1 rows) of f32, row stride = stride_elems,
// packed contiguously at LDS byte address lds_addr. D# per CDNA5 ISA §8.3/8.4.
__device__ __forceinline__ void tdm_load_2d_f32(unsigned lds_addr, const void* gaddr,
                                                unsigned d0_elems, unsigned d1,
                                                unsigned stride_elems)
{
    const unsigned long long ga = (unsigned long long)gaddr;
    v4u g0;
    g0[0] = 1u;                                   // count=1 (valid), user mode
    g0[1] = lds_addr;                             // lds_addr bits[63:32]
    g0[2] = (unsigned)(ga & 0xFFFFFFFFu);         // global_addr[31:0]
    g0[3] = (unsigned)((ga >> 32) & 0x01FFFFFFu)  // global_addr[56:32]
          | 0x80000000u;                          // type=2 ("image")

    unsigned dw0 = 2u << 16;                      // data_size=2 (4 bytes)
    unsigned dw1 = (d0_elems & 0xFFFFu) << 16;    // tensor_dim0[15:0]
    unsigned dw2 = ((d0_elems >> 16) & 0xFFFFu)   // tensor_dim0[31:16]
                 | ((d1 & 0xFFFFu) << 16);        // tensor_dim1[15:0]
    unsigned dw3 = ((d1 >> 16) & 0xFFFFu)         // tensor_dim1[31:16]
                 | ((d0_elems & 0xFFFFu) << 16);  // tile_dim0
    unsigned dw4 = (d1 & 0xFFFFu);                // tile_dim1 (tile_dim2=0)
    unsigned dw5 = stride_elems;                  // tensor_dim0_stride[31:0]
    unsigned dw6 = 0;                             // stride[47:32], dim1_stride lo
    unsigned dw7 = 0;
    v8i g1;
    g1[0] = (int)dw0; g1[1] = (int)dw1; g1[2] = (int)dw2; g1[3] = (int)dw3;
    g1[4] = (int)dw4; g1[5] = (int)dw5; g1[6] = (int)dw6; g1[7] = (int)dw7;
    v4i z4 = {0, 0, 0, 0};
#if defined(__clang_major__) && (__clang_major__ >= 23)
    v8i z8 = {0, 0, 0, 0, 0, 0, 0, 0};
    __builtin_amdgcn_tensor_load_to_lds(g0, g1, z4, z4, z8, 0);   // 6-arg form
#else
    __builtin_amdgcn_tensor_load_to_lds(g0, g1, z4, z4, 0);       // 5-arg form
#endif
}
#endif

// =====================================================================
// WMMA GEMM: C[M,NO] = epilogue(A[M,K] @ Wbf[NO,K]^T + bias)
// MODE 0: store; MODE 1: GELU(exact); MODE 2: C = relu(C + result)
// block = 128 threads (4 waves). Block tile: 80 (M) x 64 (N).
// grid = (M/80, NO/64). K multiple of 32. Weights pre-converted to bf16.
// A chunk (80x32 f32 = 10KB) staged via TDM (or async-to-LDS fallback),
// then converted once to a bf16 LDS tile consumed by 5 WMMAs / wave.
// =====================================================================
template<int MODE>
__global__ __launch_bounds__(128)
void gemm_wmma_kernel(const float* __restrict__ A, const __bf16* __restrict__ W,
                      const float* __restrict__ bias, float* __restrict__ C,
                      int K, int NO)
{
    __shared__ __align__(16) float  sAf[MT_ * 32];   // 10 KB staged f32 chunk
    __shared__ __align__(16) __bf16 sAb[MT_ * 32];   // 5 KB bf16 tile

    const int tid  = threadIdx.x;
    const int wave = tid >> 5;
    const int lane = tid & 31;
    const int m0   = blockIdx.x * MT_;
    const int n0   = (blockIdx.y * 4 + wave) * 16;
    const int half = lane >> 4;       // lane half-group
    const int r    = lane & 15;       // row (A) / col (B) within 16-tile
    const int kb0  = half * 8;        // K sub-chunk 0 of this half-group
    const int kb1  = 16 + half * 8;   // K sub-chunk 1

    const __bf16*  wrow = W + (size_t)(n0 + r) * K;
    const unsigned ldsA = (unsigned)(size_t)(lds_f32*)sAf;

    v8f acc[MSUB_];
#pragma unroll
    for (int s = 0; s < MSUB_; ++s) acc[s] = (v8f){};

    for (int kk = 0; kk < K; kk += 32) {
        // ---- stage A chunk (80 rows x 32 f32) into LDS ----
#if __has_builtin(__builtin_amdgcn_tensor_load_to_lds)
        if (wave == 0) {
            tdm_load_2d_f32(ldsA, A + (size_t)m0 * K + kk, 32u, (unsigned)MT_,
                            (unsigned)K);
            __builtin_amdgcn_s_wait_tensorcnt(0);
        }
#else
        // per-lane async copy fallback: 640 x b128; id -> row=id/8, col4=(id%8)*4
#pragma unroll
        for (int i = 0; i < MSUB_; ++i) {
            const int id  = i * 128 + tid;
            const int row = id >> 3;
            const int c4  = (id & 7) * 4;
            const float* g = A + (size_t)(m0 + row) * K + kk + c4;
            const unsigned laddr = ldsA + (unsigned)(id * 16);
            asm volatile("global_load_async_to_lds_b128 %0, %1, off"
                         :: "v"(laddr), "v"(g) : "memory");
        }
        wait_asynccnt0();
#endif
        __syncthreads();

        // ---- one-shot cooperative f32 -> bf16 conversion ----
#pragma unroll
        for (int i = 0; i < MSUB_; ++i) {
            const int id = i * 128 + tid;
            float4 fv = *(const float4*)&sAf[id * 4];
            __bf16* d = &sAb[id * 4];
            d[0] = (__bf16)fv.x; d[1] = (__bf16)fv.y;
            d[2] = (__bf16)fv.z; d[3] = (__bf16)fv.w;
        }
        if (kk + 32 < K) __builtin_prefetch(wrow + kk + 32, 0, 0);
        __syncthreads();

        // ---- B fragment: bf16 row (n0+r) of W, ISA half-lane K split ----
        v8bf blo = *(const v8bf*)(wrow + kk + kb0);
        v8bf bhi = *(const v8bf*)(wrow + kk + kb1);
        v16bf bfrag;
#pragma unroll
        for (int i = 0; i < 8; ++i) { bfrag[i] = blo[i]; bfrag[8 + i] = bhi[i]; }

        // ---- 5 M sub-tiles: bf16 A fragments straight from LDS ----
#pragma unroll
        for (int s = 0; s < MSUB_; ++s) {
            const __bf16* ar = &sAb[(s * 16 + r) * 32];
            v8bf alo = *(const v8bf*)(ar + kb0);
            v8bf ahi = *(const v8bf*)(ar + kb1);
            v16bf afrag;
#pragma unroll
            for (int i = 0; i < 8; ++i) { afrag[i] = alo[i]; afrag[8 + i] = ahi[i]; }

            acc[s] = __builtin_amdgcn_wmma_f32_16x16x32_bf16(
                         false, afrag, false, bfrag, (short)0, acc[s],
                         false, false);
        }
        __syncthreads();
    }

    // ---- epilogue: lane = column n0+r, VGPR v = row +half*8+v ----
    const int   ccol = n0 + r;
    const float bv   = bias[ccol];
#pragma unroll
    for (int s = 0; s < MSUB_; ++s) {
        const int crow0 = m0 + s * 16 + half * 8;
#pragma unroll
        for (int v = 0; v < 8; ++v) {
            size_t idx = (size_t)(crow0 + v) * NO + ccol;
            float val = acc[s][v] + bv;
            if (MODE == 1) {
                val = 0.5f * val * (1.0f + erff(val * 0.70710678118654752f));
            } else if (MODE == 2) {
                val += C[idx];
                val = fmaxf(val, 0.0f);
            }
            C[idx] = val;
        }
    }
}

// =====================================================================
// weights f32 -> bf16 one-shot conversion
// =====================================================================
__global__ void cvt_bf16_kernel(const float* __restrict__ s, __bf16* __restrict__ d,
                                int n)
{
    const int i = blockIdx.x * 256 + threadIdx.x;
    if (i < n) d[i] = (__bf16)s[i];
}

// =====================================================================
// Per-sequence attention: T1=13 tokens, 4 heads (one wave each), hd=32
// qkv: [BN, T1, 384] (q|k|v), o: [BN, T1, 128]
// Q/K/V staged into LDS with GLOBAL_LOAD_ASYNC_TO_LDS_B32.
// =====================================================================
__global__ __launch_bounds__(128)
void attn_kernel(const float* __restrict__ qkv, float* __restrict__ o)
{
    __shared__ float sQ[NH_][T1_ * 32];
    __shared__ float sK[NH_][T1_ * 32];
    __shared__ float sV[NH_][T1_ * 32];

    const int seq  = blockIdx.x;
    const int head = threadIdx.x >> 5;
    const int lane = threadIdx.x & 31;

    const float* base = qkv + (size_t)seq * T1_ * (3 * H_) + head * 32 + lane;
    const unsigned qb = (unsigned)(size_t)(lds_f32*)&sQ[head][0];
    const unsigned kb = (unsigned)(size_t)(lds_f32*)&sK[head][0];
    const unsigned vb = (unsigned)(size_t)(lds_f32*)&sV[head][0];
#pragma unroll
    for (int t = 0; t < T1_; ++t) {
        const unsigned loff = (unsigned)((t * 32 + lane) * 4);
        const float* gq = base + t * (3 * H_);
        const float* gk = gq + H_;
        const float* gv = gq + 2 * H_;
        asm volatile("global_load_async_to_lds_b32 %0, %1, off"
                     :: "v"(qb + loff), "v"(gq) : "memory");
        asm volatile("global_load_async_to_lds_b32 %0, %1, off"
                     :: "v"(kb + loff), "v"(gk) : "memory");
        asm volatile("global_load_async_to_lds_b32 %0, %1, off"
                     :: "v"(vb + loff), "v"(gv) : "memory");
    }
    wait_asynccnt0();
    __syncthreads();

    if (lane < T1_) {
        float s[T1_];
        float m = -1e30f;
#pragma unroll
        for (int j = 0; j < T1_; ++j) {
            float d = 0.0f;
#pragma unroll
            for (int c = 0; c < 32; ++c)
                d += sQ[head][lane * 32 + c] * sK[head][j * 32 + c];
            s[j] = d * 0.17677669529663687f;   // 1/sqrt(32)
            m = fmaxf(m, s[j]);
        }
        float sum = 0.0f;
#pragma unroll
        for (int j = 0; j < T1_; ++j) { s[j] = __expf(s[j] - m); sum += s[j]; }
        const float inv = 1.0f / sum;
        float* op = o + ((size_t)seq * T1_ + lane) * H_ + head * 32;
#pragma unroll
        for (int c = 0; c < 32; ++c) {
            float a = 0.0f;
#pragma unroll
            for (int j = 0; j < T1_; ++j) a += s[j] * sV[head][j * 32 + c];
            op[c] = a * inv;
        }
    }
}

// =====================================================================
// h = LayerNorm(h + r) * g + b      (one token per block, 128 threads)
// =====================================================================
__global__ __launch_bounds__(128)
void add_ln_kernel(float* __restrict__ h, const float* __restrict__ r,
                   const float* __restrict__ g, const float* __restrict__ b)
{
    __shared__ float sred[8];
    const size_t base = (size_t)blockIdx.x * H_;
    const int f = threadIdx.x;
    float x = h[base + f] + r[base + f];
    float s1 = x, s2 = x * x;
#pragma unroll
    for (int off = 16; off > 0; off >>= 1) {
        s1 += __shfl_xor(s1, off, 32);
        s2 += __shfl_xor(s2, off, 32);
    }
    const int wave = f >> 5, lane = f & 31;
    if (lane == 0) { sred[wave] = s1; sred[4 + wave] = s2; }
    __syncthreads();
    const float sum = sred[0] + sred[1] + sred[2] + sred[3];
    const float sq  = sred[4] + sred[5] + sred[6] + sred[7];
    const float mean = sum * (1.0f / H_);
    const float var  = sq * (1.0f / H_) - mean * mean;
    h[base + f] = (x - mean) * rsqrtf(var + EPS_) * g[f] + b[f];
}

// =====================================================================
// fused input projection + LayerNorm + step embedding
// =====================================================================
__global__ __launch_bounds__(128)
void embed_kernel(const float* __restrict__ Xk, const int* __restrict__ kidx,
                  const float* __restrict__ temb, const float* __restrict__ semb,
                  const float* __restrict__ w, const float* __restrict__ bb,
                  const float* __restrict__ g, const float* __restrict__ be,
                  float* __restrict__ h)
{
    __shared__ float sx[XIN_];
    __shared__ float sred[8];
    const size_t tok = blockIdx.x;
    const int t    = (int)(tok % T1_);
    const int bidx = (int)(tok / ((size_t)N_ * T1_));
    const int f = threadIdx.x;

    if (f < CIN_)       sx[f] = Xk[tok * CIN_ + f];
    else if (f < XIN_)  sx[f] = temb[t * TED_ + (f - CIN_)];
    __syncthreads();

    const float* wr = w + f * XIN_;
    float acc = bb[f];
#pragma unroll
    for (int j = 0; j < XIN_; ++j) acc += wr[j] * sx[j];

    float s1 = acc, s2 = acc * acc;
#pragma unroll
    for (int off = 16; off > 0; off >>= 1) {
        s1 += __shfl_xor(s1, off, 32);
        s2 += __shfl_xor(s2, off, 32);
    }
    const int wave = f >> 5, lane = f & 31;
    if (lane == 0) { sred[wave] = s1; sred[4 + wave] = s2; }
    __syncthreads();
    const float sum = sred[0] + sred[1] + sred[2] + sred[3];
    const float sq  = sred[4] + sred[5] + sred[6] + sred[7];
    const float mean = sum * (1.0f / H_);
    const float var  = sq * (1.0f / H_) - mean * mean;
    float y = (acc - mean) * rsqrtf(var + EPS_) * g[f] + be[f];
    y += semb[(size_t)kidx[bidx] * H_ + f];
    h[tok * H_ + f] = y;
}

// =====================================================================
// graph: degree count, edge weights, atomic neighbour scatter
// =====================================================================
__global__ void deg_kernel(const int* __restrict__ ei, float* __restrict__ deg)
{
    const int e = blockIdx.x * 256 + threadIdx.x;
    if (e >= EN_) return;
    const int r = (e < E_) ? ei[e] : (e - E_);
    atomicAdd(&deg[r], 1.0f);
}

__global__ void ew_kernel(const int* __restrict__ ei, const float* __restrict__ deg,
                          float* __restrict__ ew)
{
    const int e = blockIdx.x * 256 + threadIdx.x;
    if (e >= EN_) return;
    const int r = (e < E_) ? ei[e] : (e - E_);
    float d = deg[r];
    if (d < 1.0f) d = 1.0f;
    ew[e] = 1.0f / d;
}

__global__ __launch_bounds__(128)
void scatter_kernel(const float* __restrict__ h, const int* __restrict__ ei,
                    const float* __restrict__ ew, float* __restrict__ neigh)
{
    const int e  = blockIdx.x;
    const int ch = threadIdx.x;
    int r, c;
    if (e < E_) { r = ei[e]; c = ei[E_ + e]; } else { r = e - E_; c = r; }
    const float w = ew[e];
#pragma unroll
    for (int b = 0; b < B_; ++b)
        for (int t = 0; t < T1_; ++t) {
            const size_t si = (((size_t)b * N_ + c) * T1_ + t) * H_ + ch;
            const size_t di = (((size_t)b * N_ + r) * T1_ + t) * H_ + ch;
            atomicAdd(&neigh[di], w * h[si]);
        }
}

// =====================================================================
// final 128 -> 8 projection + softmax (one token per 32-thread block)
// =====================================================================
__global__ __launch_bounds__(32)
void out_kernel(const float* __restrict__ h, const float* __restrict__ w,
                const float* __restrict__ b, float* __restrict__ out)
{
    const size_t tok = blockIdx.x;
    const int lane = threadIdx.x;
    float logit = -1e30f;
    if (lane < SD_) {
        float acc = b[lane];
        const float* hr = h + tok * H_;
        const float* wr = w + lane * H_;
#pragma unroll
        for (int j = 0; j < H_; ++j) acc += hr[j] * wr[j];
        logit = acc;
    }
    float m = logit;
#pragma unroll
    for (int off = 4; off > 0; off >>= 1) m = fmaxf(m, __shfl_xor(m, off, 8));
    float p = (lane < SD_) ? __expf(logit - m) : 0.0f;
    float s = p;
#pragma unroll
    for (int off = 4; off > 0; off >>= 1) s += __shfl_xor(s, off, 8);
    if (lane < SD_) out[tok * SD_ + lane] = p / s;
}

// =====================================================================
// host orchestration
// =====================================================================
extern "C" void kernel_launch(void* const* d_in, const int* in_sizes, int n_in,
                              void* d_out, int out_size, void* d_ws, size_t ws_size,
                              hipStream_t stream)
{
    const float* X_k  = (const float*)d_in[0];
    const int*   kidx = (const int*)  d_in[1];
    const int*   ei   = (const int*)  d_in[2];
    const float* temb = (const float*)d_in[3];
    const float* semb = (const float*)d_in[4];
    const float* ipw  = (const float*)d_in[5];
    const float* ipb  = (const float*)d_in[6];
    const float* ing  = (const float*)d_in[7];
    const float* inb  = (const float*)d_in[8];
    const float* qkvw = (const float*)d_in[9];
    const float* qkvb = (const float*)d_in[10];
    const float* aow  = (const float*)d_in[11];
    const float* aob  = (const float*)d_in[12];
    const float* tng  = (const float*)d_in[13];
    const float* tnb  = (const float*)d_in[14];
    const float* f1w  = (const float*)d_in[15];
    const float* f1b  = (const float*)d_in[16];
    const float* f2w  = (const float*)d_in[17];
    const float* f2b  = (const float*)d_in[18];
    const float* fng  = (const float*)d_in[19];
    const float* fnb  = (const float*)d_in[20];
    const float* gsw  = (const float*)d_in[21];
    const float* gsb  = (const float*)d_in[22];
    const float* gnw  = (const float*)d_in[23];
    const float* gnb  = (const float*)d_in[24];
    const float* glng = (const float*)d_in[25];
    const float* glnb = (const float*)d_in[26];
    const float* outw = (const float*)d_in[27];
    const float* outb = (const float*)d_in[28];

    char* ws = (char*)d_ws;
    size_t off = 0;
    auto alloc = [&](size_t bytes) {
        void* p = ws + off;
        off += (bytes + 255) & ~(size_t)255;
        return p;
    };
    float*  h     = (float*) alloc((size_t)TOK_ * H_ * 4);
    float*  bufQ  = (float*) alloc((size_t)TOK_ * 3 * H_ * 4);   // qkv / ffn1
    float*  bufA  = (float*) alloc((size_t)TOK_ * H_ * 4);       // attn output
    float*  bufB2 = (float*) alloc((size_t)TOK_ * H_ * 4);       // residual branch
    float*  neigh = (float*) alloc((size_t)TOK_ * H_ * 4);
    float*  deg   = (float*) alloc((size_t)N_ * 4);
    float*  ew    = (float*) alloc((size_t)EN_ * 4);
    // bf16 weight copies
    const int nqkv = L_ * 3 * H_ * H_;
    const int nao  = L_ * H_ * H_;
    const int nf1  = L_ * FF_ * H_;
    const int nf2  = L_ * H_ * FF_;
    const int ngs  = L_ * H_ * H_;
    const int ngn  = L_ * H_ * H_;
    __bf16* qkvwB = (__bf16*)alloc((size_t)nqkv * 2);
    __bf16* aowB  = (__bf16*)alloc((size_t)nao  * 2);
    __bf16* f1wB  = (__bf16*)alloc((size_t)nf1  * 2);
    __bf16* f2wB  = (__bf16*)alloc((size_t)nf2  * 2);
    __bf16* gswB  = (__bf16*)alloc((size_t)ngs  * 2);
    __bf16* gnwB  = (__bf16*)alloc((size_t)ngn  * 2);

    // ---- one-shot weight conversion to bf16 ----
    cvt_bf16_kernel<<<(nqkv + 255) / 256, 256, 0, stream>>>(qkvw, qkvwB, nqkv);
    cvt_bf16_kernel<<<(nao  + 255) / 256, 256, 0, stream>>>(aow,  aowB,  nao);
    cvt_bf16_kernel<<<(nf1  + 255) / 256, 256, 0, stream>>>(f1w,  f1wB,  nf1);
    cvt_bf16_kernel<<<(nf2  + 255) / 256, 256, 0, stream>>>(f2w,  f2wB,  nf2);
    cvt_bf16_kernel<<<(ngs  + 255) / 256, 256, 0, stream>>>(gsw,  gswB,  ngs);
    cvt_bf16_kernel<<<(ngn  + 255) / 256, 256, 0, stream>>>(gnw,  gnwB,  ngn);

    // ---- graph preprocessing ----
    (void)hipMemsetAsync(deg, 0, (size_t)N_ * 4, stream);
    deg_kernel<<<(EN_ + 255) / 256, 256, 0, stream>>>(ei, deg);
    ew_kernel<<<(EN_ + 255) / 256, 256, 0, stream>>>(ei, deg, ew);

    // ---- input embedding ----
    embed_kernel<<<TOK_, 128, 0, stream>>>(X_k, kidx, temb, semb,
                                           ipw, ipb, ing, inb, h);

    const dim3 g384(TOK_ / MT_, (3 * H_) / 64);
    const dim3 g256(TOK_ / MT_, FF_ / 64);
    const dim3 g128(TOK_ / MT_, H_ / 64);

    for (int l = 0; l < L_; ++l) {
        // --- temporal attention ---
        gemm_wmma_kernel<0><<<g384, 128, 0, stream>>>(
            h, qkvwB + (size_t)l * 3 * H_ * H_, qkvb + (size_t)l * 3 * H_,
            bufQ, H_, 3 * H_);
        attn_kernel<<<B_ * N_, 128, 0, stream>>>(bufQ, bufA);
        gemm_wmma_kernel<0><<<g128, 128, 0, stream>>>(
            bufA, aowB + (size_t)l * H_ * H_, aob + (size_t)l * H_,
            bufB2, H_, H_);
        add_ln_kernel<<<TOK_, 128, 0, stream>>>(h, bufB2,
            tng + (size_t)l * H_, tnb + (size_t)l * H_);

        // --- FFN ---
        gemm_wmma_kernel<1><<<g256, 128, 0, stream>>>(
            h, f1wB + (size_t)l * FF_ * H_, f1b + (size_t)l * FF_,
            bufQ, H_, FF_);
        gemm_wmma_kernel<0><<<g128, 128, 0, stream>>>(
            bufQ, f2wB + (size_t)l * H_ * FF_, f2b + (size_t)l * H_,
            bufB2, FF_, H_);
        add_ln_kernel<<<TOK_, 128, 0, stream>>>(h, bufB2,
            fng + (size_t)l * H_, fnb + (size_t)l * H_);

        // --- graph gate ---
        (void)hipMemsetAsync(neigh, 0, (size_t)TOK_ * H_ * 4, stream);
        scatter_kernel<<<EN_, 128, 0, stream>>>(h, ei, ew, neigh);
        gemm_wmma_kernel<0><<<g128, 128, 0, stream>>>(
            h, gswB + (size_t)l * H_ * H_, gsb + (size_t)l * H_,
            bufB2, H_, H_);
        gemm_wmma_kernel<2><<<g128, 128, 0, stream>>>(
            neigh, gnwB + (size_t)l * H_ * H_, gnb + (size_t)l * H_,
            bufB2, H_, H_);
        add_ln_kernel<<<TOK_, 128, 0, stream>>>(h, bufB2,
            glng + (size_t)l * H_, glnb + (size_t)l * H_);
    }

    out_kernel<<<TOK_, 32, 0, stream>>>(h, outw, outb, (float*)d_out);
}
